// CSPNetLight_11192684773497
// MI455X (gfx1250) — matrix operand order repaired and checked by reading, not compiled, not run on hardware
//
#include <hip/hip_runtime.h>
#include <hip/hip_bf16.h>

#define BS 16
#define NM 100
#define HDIM 128
#define TDIM 256
#define NTYPES 100
#define NFREQ 10
#define NLAYERS 4
#define NNODES (BS * NM) // 1600
#define NTILES (NNODES / 16) // 100

typedef _Float16 v16h __attribute__((ext_vector_type(16)));
typedef _Float16 v8h  __attribute__((ext_vector_type(8)));
typedef float    v8f  __attribute__((ext_vector_type(8)));

// Fast silu: x * sigmoid(x) using the native v_rcp_f32 (1-ulp) — avoids the
// IEEE v_div_scale/Newton chain the compiler emits for '/' and __fdividef.
__device__ __forceinline__ float silu_f(float x) {
  return x * __builtin_amdgcn_rcpf(1.0f + __expf(-x));
}

__device__ __forceinline__ v8f wmma16(v16h a, v16h b, v8f c) {
  // D = A(16x32 f16) * B(32x16 f16) + C(16x16 f32)
  return __builtin_amdgcn_wmma_f32_16x16x32_f16(false, a, false, b, (short)0, c,
                                                false, false);
}

// Pack an A-fragment (16x32 f16) from a row-major [M][K] tile row (LDS or global).
// Per CDNA5 ISA 7.12.2: lane<16 holds K {0..7, 16..23}; lane>=16 holds {8..15, 24..31}.
// 'off' = 32*ktile + 8*(lane>=16).
__device__ __forceinline__ v16h pack_afrag(const _Float16* rowp, int off) {
  v8h lo = *(const v8h*)(rowp + off);
  v8h hi = *(const v8h*)(rowp + off + 16);
  v16h a;
#pragma unroll
  for (int k = 0; k < 8; ++k) { a[k] = lo[k]; a[8 + k] = hi[k]; }
  return a;
}

// ---------------------------------------------------------------------------
// Fused edge kernel (the ~95%-of-FLOPs path). One (b,i) node per workgroup,
// 8 waves; wave w owns H columns [16w,16w+16).  Per 16-row j tile:
//   feat(16x64 f16 sinusoids) --2x WMMA--> eproj slab
//   m1 = silu(eproj + xj[j] + xi[i] + gproj[b] + b1) -> LDS f16
//   m2 slab = m1 @ W2  --4x WMMA-->  silu, masked mean over j in registers
// NOTE: xj is over-allocated by 16 rows so the last tile's dummy rows can be
// read unclamped (finite garbage, masked before the mean) -> the 8 epilogue
// loads become immediate-offset loads off one base address.
// ---------------------------------------------------------------------------
__global__ __launch_bounds__(256) void k_edge(
    const float* __restrict__ xj, const float* __restrict__ xi,
    const float* __restrict__ gproj, const float* __restrict__ b1,
    const float* __restrict__ b2, const _Float16* __restrict__ W1e, // [64][128]
    const _Float16* __restrict__ W2,                                 // [128][128]
    const float* __restrict__ frac, _Float16* __restrict__ msg) {
  const int wg = blockIdx.x;
  const int b = wg / NM;
  const int i = wg % NM;
  const int tid = threadIdx.x;
  const int wave = tid >> 5;
  const int lane = tid & 31;
  const int group = lane >> 4;
  const int arow = lane & 15;
  const int ncol = wave * 16;
  const int h = ncol + (lane & 15);

  __shared__ __align__(16) _Float16 F[16][64];    // feature tile (K 60 -> 64)
  __shared__ __align__(16) _Float16 M1[16][HDIM]; // silu(m1) tile, f16

  const float bias1 = xi[(b * NM + i) * HDIM + h] + gproj[b * HDIM + h] + b1[h];
  const float bias2 = b2[h];
  float fi[3];
#pragma unroll
  for (int c = 0; c < 3; ++c) fi[c] = frac[(b * NM + i) * 3 + c];

  // Preload B fragments once: lane = N column (lane&15), 16 K per lane in A's table order.
  v16h bw1[2], bw2[4];
#pragma unroll
  for (int t = 0; t < 2; ++t)
    for (int e = 0; e < 16; ++e)
      bw1[t][e] = W1e[(t * 32 + group * 8 + e + (e & 8)) * HDIM + h];
#pragma unroll
  for (int t = 0; t < 4; ++t)
    for (int e = 0; e < 16; ++e)
      bw2[t][e] = W2[(t * 32 + group * 8 + e + (e & 8)) * HDIM + h];

  // Base pointer for this lane's xj column slice (advances by 16 rows/tile).
  const float* xjp = xj + (size_t)(b * NM + group * 8) * HDIM + h;

  float sum = 0.0f;

  for (int j0 = 0; j0 < NM; j0 += 16) {
    // Prefetch next tile's xj row (gfx1250 global_prefetch path).
    __builtin_prefetch(xjp + (j0 + 16) * HDIM, 0, 0);

    // Feature tile via sincos: 512 work items, 2 per thread; each produces
    // both the sin (k) and cos (k+30) feature, or zero-pads K=60..63.
    for (int e = tid; e < 16 * 32; e += 256) {
      const int m = e >> 5;
      const int k2 = e & 31;
      int jj = j0 + m;
      if (jj > NM - 1) jj = NM - 1; // clamped dummy rows, masked later
      if (k2 < 30) {
        const int c = k2 / NFREQ;
        const int q = k2 % NFREQ;
        const float fd = frac[(b * NM + jj) * 3 + c] - fi[c];
        const float ang = fd * (6.28318530717958647692f * (float)q);
        float sv, cv;
        __sincosf(ang, &sv, &cv);
        F[m][k2] = (_Float16)sv;
        F[m][k2 + 30] = (_Float16)cv;
      } else {
        const int z = k2 - 30; // 0 or 1
        F[m][60 + z] = (_Float16)0.0f;
        F[m][62 + z] = (_Float16)0.0f;
      }
    }
    __syncthreads();

    // GEMM1: eproj slab = feat(16x64) @ W1e(64x16 slab)
    v8f acc;
#pragma unroll
    for (int r = 0; r < 8; ++r) acc[r] = 0.0f;
#pragma unroll
    for (int t = 0; t < 2; ++t)
      acc = wmma16(pack_afrag(&F[arow][0], t * 32 + group * 8), bw1[t], acc);

    // m1 = silu(acc + xj + bias1) -> LDS f16 (C/D layout: M = r + 8*group)
#pragma unroll
    for (int r = 0; r < 8; ++r) {
      const int m = r + group * 8;
      const float v = acc[r] + xjp[(j0 + r) * HDIM] + bias1;
      M1[m][h] = (_Float16)silu_f(v);
    }
    __syncthreads();

    // GEMM2: m2 slab = m1(16x128) @ W2(128x16 slab)
    v8f acc2;
#pragma unroll
    for (int r = 0; r < 8; ++r) acc2[r] = 0.0f;
#pragma unroll
    for (int t = 0; t < 4; ++t)
      acc2 = wmma16(pack_afrag(&M1[arow][0], t * 32 + group * 8), bw2[t], acc2);

#pragma unroll
    for (int r = 0; r < 8; ++r) {
      const int m = r + group * 8;
      if (j0 + m < NM) sum += silu_f(acc2[r] + bias2);
    }
    __syncthreads();
  }

  sum += __shfl_xor(sum, 16);
  if (lane < 16)
    msg[(b * NM + i) * HDIM + h] = (_Float16)(sum * (1.0f / (float)NM));
}

// ---------------------------------------------------------------------------
// WMMA node GEMMs: 100 M-tiles of 16 node rows; 8 waves per block.
// A fragments loaded straight from global f16 activations (L2-resident).
// ---------------------------------------------------------------------------

// [xj | xi] = xn @ Wx, Wx = [128][256] ( [:,0:128]=W1[0:128], [:,128:256]=W1[128:256] )
__global__ __launch_bounds__(256) void k_xproj_wmma(
    const _Float16* __restrict__ A, const _Float16* __restrict__ Bx,
    float* __restrict__ xj, float* __restrict__ xi) {
  const int tile = blockIdx.x;
  const int wave = threadIdx.x >> 5, lane = threadIdx.x & 31;
  const int group = lane >> 4, arow = lane & 15;
  const _Float16* Arow = A + (tile * 16 + arow) * HDIM;
  v16h af[4];
#pragma unroll
  for (int t = 0; t < 4; ++t) af[t] = pack_afrag(Arow, t * 32 + group * 8);
  const int hh = wave * 16 + (lane & 15);
#pragma unroll
  for (int s = 0; s < 2; ++s) {
    const int ncol = s * HDIM + wave * 16;
    v8f acc;
#pragma unroll
    for (int r = 0; r < 8; ++r) acc[r] = 0.0f;
#pragma unroll
    for (int t = 0; t < 4; ++t) {
      v16h bf;
      for (int e = 0; e < 16; ++e)
        bf[e] = Bx[(t * 32 + group * 8 + e + (e & 8)) * 256 + ncol + (lane & 15)];
      acc = wmma16(af[t], bf, acc);
    }
    float* outp = (s == 0) ? xj : xi;
#pragma unroll
    for (int r = 0; r < 8; ++r) {
      const int m = r + group * 8;
      outp[(tile * 16 + m) * HDIM + hh] = acc[r];
    }
  }
}

// h1 = silu(concat(xn, msg) @ Wm1 + b1), written as f16.
__global__ __launch_bounds__(256) void k_mlp1_wmma(
    const _Float16* __restrict__ Axn, const _Float16* __restrict__ Amsg,
    const _Float16* __restrict__ Bm1, const float* __restrict__ b1,
    _Float16* __restrict__ h1) {
  const int tile = blockIdx.x;
  const int wave = threadIdx.x >> 5, lane = threadIdx.x & 31;
  const int group = lane >> 4, arow = lane & 15;
  const int ncol = wave * 16;
  const int hh = ncol + (lane & 15);
  const _Float16* A1 = Axn + (tile * 16 + arow) * HDIM;
  const _Float16* A2 = Amsg + (tile * 16 + arow) * HDIM;
  v8f acc;
#pragma unroll
  for (int r = 0; r < 8; ++r) acc[r] = 0.0f;
#pragma unroll
  for (int t = 0; t < 4; ++t) {
    v16h bf;
    for (int e = 0; e < 16; ++e)
      bf[e] = Bm1[(t * 32 + group * 8 + e + (e & 8)) * HDIM + hh];
    acc = wmma16(pack_afrag(A1, t * 32 + group * 8), bf, acc);
  }
#pragma unroll
  for (int t = 0; t < 4; ++t) {
    v16h bf;
    for (int e = 0; e < 16; ++e)
      bf[e] = Bm1[((HDIM + t * 32) + group * 8 + e + (e & 8)) * HDIM + hh];
    acc = wmma16(pack_afrag(A2, t * 32 + group * 8), bf, acc);
  }
  const float bias = b1[hh];
#pragma unroll
  for (int r = 0; r < 8; ++r) {
    const int m = r + group * 8;
    h1[(tile * 16 + m) * HDIM + hh] = (_Float16)silu_f(acc[r] + bias);
  }
}

// x = xn + silu(h1 @ Wm2 + b2)  (residual kept fp32)
__global__ __launch_bounds__(256) void k_mlp2_wmma(
    const _Float16* __restrict__ Ah1, const _Float16* __restrict__ Bm2,
    const float* __restrict__ b2, const float* __restrict__ xn,
    float* __restrict__ x) {
  const int tile = blockIdx.x;
  const int wave = threadIdx.x >> 5, lane = threadIdx.x & 31;
  const int group = lane >> 4, arow = lane & 15;
  const int ncol = wave * 16;
  const int hh = ncol + (lane & 15);
  const _Float16* A = Ah1 + (tile * 16 + arow) * HDIM;
  v8f acc;
#pragma unroll
  for (int r = 0; r < 8; ++r) acc[r] = 0.0f;
#pragma unroll
  for (int t = 0; t < 4; ++t) {
    v16h bf;
    for (int e = 0; e < 16; ++e)
      bf[e] = Bm2[(t * 32 + group * 8 + e + (e & 8)) * HDIM + hh];
    acc = wmma16(pack_afrag(A, t * 32 + group * 8), bf, acc);
  }
  const float bias = b2[hh];
#pragma unroll
  for (int r = 0; r < 8; ++r) {
    const int m = r + group * 8;
    const int idx = (tile * 16 + m) * HDIM + hh;
    x[idx] = xn[idx] + silu_f(acc[r] + bias);
  }
}

// ---------------------------------------------------------------------------
// Small helper kernels (latency-bound, once-per-call or tiny)
// ---------------------------------------------------------------------------
__global__ void k_lat_embed(const float* __restrict__ L, float* __restrict__ out) {
  int tid = threadIdx.x;
  if (tid >= BS * 9) return;
  int b = tid / 9, r = tid % 9, i = r / 3, k = r % 3;
  float s = 0.f;
#pragma unroll
  for (int j = 0; j < 3; ++j) s += L[b * 9 + i * 3 + j] * L[b * 9 + k * 3 + j];
  out[tid] = s;
}

__global__ void k_tproj(const float* __restrict__ t_embed,
                        const float* __restrict__ node_W,
                        const float* __restrict__ node_b,
                        float* __restrict__ tproj) {
  int b = blockIdx.x, hh = threadIdx.x;
  float s = node_b[hh];
  for (int d = 0; d < TDIM; ++d)
    s += t_embed[b * TDIM + d] * node_W[(HDIM + d) * HDIM + hh];
  tproj[b * HDIM + hh] = s;
}

__global__ void k_atom_embed(const float* __restrict__ at,
                             const float* __restrict__ atom_W,
                             const float* __restrict__ atom_b,
                             float* __restrict__ ae) {
  int idx = blockIdx.x * blockDim.x + threadIdx.x;
  if (idx >= NNODES * HDIM) return;
  int n = idx >> 7, hh = idx & 127;
  float s = atom_b[hh];
  for (int t = 0; t < NTYPES; ++t) s += at[n * NTYPES + t] * atom_W[t * HDIM + hh];
  ae[idx] = s;
}

__global__ void k_node_init(const float* __restrict__ ae,
                            const float* __restrict__ node_W,
                            const float* __restrict__ tproj,
                            float* __restrict__ x) {
  int idx = blockIdx.x * blockDim.x + threadIdx.x;
  if (idx >= NNODES * HDIM) return;
  int n = idx >> 7, hh = idx & 127, b = n / NM;
  float s = tproj[b * HDIM + hh];
  const float* r = ae + n * HDIM;
#pragma unroll 4
  for (int k = 0; k < HDIM; ++k) s += r[k] * node_W[k * HDIM + hh];
  x[idx] = s;
}

// LayerNorm, one wave per row; emits fp32 and f16 copies.
__global__ __launch_bounds__(256) void k_ln(const float* __restrict__ in,
                                            float* __restrict__ out,
                                            _Float16* __restrict__ outh,
                                            const float* __restrict__ g,
                                            const float* __restrict__ bb,
                                            int rows) {
  int wave = threadIdx.x >> 5, lane = threadIdx.x & 31;
  int row = blockIdx.x * 8 + wave;
  if (row >= rows) return;
  const float* p = in + row * HDIM;
  float v[4], s = 0.f;
#pragma unroll
  for (int k = 0; k < 4; ++k) { v[k] = p[lane * 4 + k]; s += v[k]; }
#pragma unroll
  for (int off = 16; off; off >>= 1) s += __shfl_xor(s, off);
  float mu = s * (1.f / HDIM);
  float q = 0.f;
#pragma unroll
  for (int k = 0; k < 4; ++k) { float d = v[k] - mu; q += d * d; }
#pragma unroll
  for (int off = 16; off; off >>= 1) q += __shfl_xor(q, off);
  float rstd = rsqrtf(q * (1.f / HDIM) + 1e-5f);
#pragma unroll
  for (int k = 0; k < 4; ++k) {
    int hh = lane * 4 + k;
    float o = (v[k] - mu) * rstd * g[hh] + bb[hh];
    out[row * HDIM + hh] = o;
    outh[row * HDIM + hh] = (_Float16)o;
  }
}

__global__ void k_gproj(const float* __restrict__ latE, const float* __restrict__ W1,
                        float* __restrict__ gp) {
  int b = blockIdx.x, hh = threadIdx.x;
  float s = 0.f;
#pragma unroll
  for (int k = 0; k < 9; ++k) s += latE[b * 9 + k] * W1[(2 * HDIM + k) * HDIM + hh];
  gp[b * HDIM + hh] = s;
}

// Per-layer f16 weight staging: W1e[64][128], W2[128][128], Wx[128][256],
// Wm1[256][128], Wm2[128][128].
__global__ void k_wconv(const float* __restrict__ W1, const float* __restrict__ W2,
                        const float* __restrict__ M1, const float* __restrict__ M2,
                        _Float16* __restrict__ w1e, _Float16* __restrict__ w2,
                        _Float16* __restrict__ wx, _Float16* __restrict__ wm1,
                        _Float16* __restrict__ wm2) {
  int idx = blockIdx.x * blockDim.x + threadIdx.x;
  if (idx < 64 * HDIM) { // W1e (edge block, rows 265.., K padded)
    int r = idx / HDIM, c = idx % HDIM;
    w1e[idx] = (r < 60) ? (_Float16)W1[(2 * HDIM + 9 + r) * HDIM + c] : (_Float16)0.0f;
  } else if (idx < 64 * HDIM + HDIM * HDIM) {
    int j = idx - 64 * HDIM;
    w2[j] = (_Float16)W2[j];
  } else if (idx < 64 * HDIM + HDIM * HDIM + HDIM * 256) {
    int j = idx - (64 * HDIM + HDIM * HDIM);
    int k = j / 256, n = j % 256;
    wx[j] = (_Float16)((n < HDIM) ? W1[k * HDIM + n] : W1[(HDIM + k) * HDIM + (n - HDIM)]);
  } else if (idx < 64 * HDIM + HDIM * HDIM + HDIM * 256 + 256 * HDIM) {
    int j = idx - (64 * HDIM + HDIM * HDIM + HDIM * 256);
    wm1[j] = (_Float16)M1[j];
  } else if (idx < 64 * HDIM + HDIM * HDIM + HDIM * 256 + 256 * HDIM + HDIM * HDIM) {
    int j = idx - (64 * HDIM + HDIM * HDIM + HDIM * 256 + 256 * HDIM);
    wm2[j] = (_Float16)M2[j];
  }
}

__global__ void k_meanx(const float* __restrict__ xn, float* __restrict__ mx) {
  int b = blockIdx.x, hh = threadIdx.x;
  float s = 0.f;
  for (int a = 0; a < NM; ++a) s += xn[(b * NM + a) * HDIM + hh];
  mx[b * HDIM + hh] = s * (1.0f / (float)NM);
}

__global__ void k_lat_head(const float* __restrict__ mx, const float* __restrict__ latW,
                           const float* __restrict__ latb, const float* __restrict__ L,
                           float* __restrict__ out) {
  __shared__ float lp[9];
  int b = blockIdx.x, tid = threadIdx.x;
  if (tid < 9) {
    float s = latb[tid];
    for (int k = 0; k < HDIM; ++k) s += mx[b * HDIM + k] * latW[k * 9 + tid];
    lp[tid] = s;
  }
  __syncthreads();
  if (tid < 9) {
    int i = tid / 3, k2 = tid % 3;
    float s = 0.f;
#pragma unroll
    for (int j = 0; j < 3; ++j) s += lp[i * 3 + j] * L[b * 9 + j * 3 + k2];
    out[b * 9 + tid] = s;
  }
}

__global__ void k_frac_head(const float* __restrict__ xn, const float* __restrict__ fW,
                            const float* __restrict__ fb, float* __restrict__ out) {
  int idx = blockIdx.x * blockDim.x + threadIdx.x;
  if (idx >= NNODES * 3) return;
  int n = idx / 3, c = idx % 3;
  float s = fb[c];
  for (int k = 0; k < HDIM; ++k) s += xn[n * HDIM + k] * fW[k * 3 + c];
  out[idx] = s;
}

__global__ void k_atype_head(const float* __restrict__ xn, const float* __restrict__ aW,
                             float* __restrict__ out) {
  int idx = blockIdx.x * blockDim.x + threadIdx.x;
  if (idx >= NNODES * NTYPES) return;
  int n = idx / NTYPES, c = idx % NTYPES;
  float s = 0.f;
#pragma unroll 4
  for (int k = 0; k < HDIM; ++k) s += xn[n * HDIM + k] * aW[k * NTYPES + c];
  out[idx] = s;
}

// ---------------------------------------------------------------------------
extern "C" void kernel_launch(void* const* d_in, const int* in_sizes, int n_in,
                              void* d_out, int out_size, void* d_ws, size_t ws_size,
                              hipStream_t stream) {
  (void)in_sizes; (void)n_in; (void)out_size; (void)ws_size;
  const float* t_embed    = (const float*)d_in[1];
  const float* atom_types = (const float*)d_in[2];
  const float* frac       = (const float*)d_in[3];
  const float* lattice    = (const float*)d_in[4];
  const float* atom_W     = (const float*)d_in[7];
  const float* atom_b     = (const float*)d_in[8];
  const float* node_W     = (const float*)d_in[9];
  const float* node_b     = (const float*)d_in[10];
  const float* fin_g  = (const float*)d_in[51];
  const float* fin_b  = (const float*)d_in[52];
  const float* lat_W  = (const float*)d_in[53];
  const float* lat_b  = (const float*)d_in[54];
  const float* frac_W = (const float*)d_in[55];
  const float* frac_b = (const float*)d_in[56];
  const float* atype_W= (const float*)d_in[57];

  char* base = (char*)d_ws;
  size_t off = 0;
  auto alloc = [&](size_t bytes) -> void* {
    void* p = base + off;
    off = (off + bytes + 255) & ~(size_t)255;
    return p;
  };
  const size_t NH = (size_t)NNODES * HDIM;
  const size_t WPL = 64 * HDIM + HDIM * HDIM + HDIM * 256 + 256 * HDIM + HDIM * HDIM;
  float* x     = (float*)alloc(NH * 4);
  float* xn    = (float*)alloc(NH * 4);
  float* xj    = (float*)alloc((NH + 16 * HDIM) * 4); // +16 pad rows (unclamped tail reads)
  float* xi    = (float*)alloc(NH * 4);
  float* ae    = (float*)alloc(NH * 4);
  float* tproj = (float*)alloc(BS * HDIM * 4);
  float* latE  = (float*)alloc(BS * 9 * 4);
  float* gp    = (float*)alloc(BS * HDIM * 4);
  float* mx    = (float*)alloc(BS * HDIM * 4);
  _Float16* xnh  = (_Float16*)alloc(NH * 2);
  _Float16* msgh = (_Float16*)alloc(NH * 2);
  _Float16* h1h  = (_Float16*)alloc(NH * 2);
  _Float16* wbuf = (_Float16*)alloc(NLAYERS * WPL * 2);

  const int TPB = 256;
  const int gNH = (int)((NH + TPB - 1) / TPB);

  // Graph embeds + node init
  k_lat_embed<<<1, 256, 0, stream>>>(lattice, latE);
  k_tproj<<<BS, HDIM, 0, stream>>>(t_embed, node_W, node_b, tproj);
  k_atom_embed<<<gNH, TPB, 0, stream>>>(atom_types, atom_W, atom_b, ae);
  k_node_init<<<gNH, TPB, 0, stream>>>(ae, node_W, tproj, x);

  // Per-layer f16 weight staging
  for (int l = 0; l < NLAYERS; ++l) {
    const int bi = 11 + l * 10;
    _Float16* wl = wbuf + (size_t)l * WPL;
    _Float16* w1e = wl;
    _Float16* w2  = w1e + 64 * HDIM;
    _Float16* wx  = w2 + HDIM * HDIM;
    _Float16* wm1 = wx + HDIM * 256;
    _Float16* wm2 = wm1 + 256 * HDIM;
    k_wconv<<<(int)((WPL + TPB - 1) / TPB), TPB, 0, stream>>>(
        (const float*)d_in[bi + 2], (const float*)d_in[bi + 4],
        (const float*)d_in[bi + 6], (const float*)d_in[bi + 8],
        w1e, w2, wx, wm1, wm2);
  }

  for (int l = 0; l < NLAYERS; ++l) {
    const int bi = 11 + l * 10;
    const float* ln_g  = (const float*)d_in[bi + 0];
    const float* ln_b  = (const float*)d_in[bi + 1];
    const float* mp_W1 = (const float*)d_in[bi + 2];
    const float* mp_b1 = (const float*)d_in[bi + 3];
    const float* mp_b2 = (const float*)d_in[bi + 5];
    const float* ml_b1 = (const float*)d_in[bi + 7];
    const float* ml_b2 = (const float*)d_in[bi + 9];
    _Float16* wl = wbuf + (size_t)l * WPL;
    _Float16* w1e = wl;
    _Float16* w2  = w1e + 64 * HDIM;
    _Float16* wx  = w2 + HDIM * HDIM;
    _Float16* wm1 = wx + HDIM * 256;
    _Float16* wm2 = wm1 + 256 * HDIM;

    k_ln<<<NNODES / 8, 256, 0, stream>>>(x, xn, xnh, ln_g, ln_b, NNODES);
    k_xproj_wmma<<<NTILES, 256, 0, stream>>>(xnh, wx, xj, xi);
    k_gproj<<<BS, HDIM, 0, stream>>>(latE, mp_W1, gp);
    k_edge<<<NNODES, 256, 0, stream>>>(xj, xi, gp, mp_b1, mp_b2, w1e, w2, frac,
                                       msgh);
    k_mlp1_wmma<<<NTILES, 256, 0, stream>>>(xnh, msgh, wm1, ml_b1, h1h);
    k_mlp2_wmma<<<NTILES, 256, 0, stream>>>(h1h, wm2, ml_b2, xn, x);
  }

  // Final LN + heads
  k_ln<<<NNODES / 8, 256, 0, stream>>>(x, xn, xnh, fin_g, fin_b, NNODES);
  k_meanx<<<BS, HDIM, 0, stream>>>(xn, mx);

  float* out = (float*)d_out;
  k_lat_head<<<BS, 16, 0, stream>>>(mx, lat_W, lat_b, lattice, out);
  k_frac_head<<<(NNODES * 3 + TPB - 1) / TPB, TPB, 0, stream>>>(xn, frac_W, frac_b,
                                                                out + BS * 9);
  k_atype_head<<<(NNODES * NTYPES + TPB - 1) / TPB, TPB, 0, stream>>>(
      xn, atype_W, out + BS * 9 + NNODES * 3);
}
